// SCENE_12386685682352
// MI455X (gfx1250) — compile-verified, matrix operand order checked by reading, not run on hardware
//
#include <hip/hip_runtime.h>

#define HEADS 4
constexpr int NN = 10000;
constexpr int EE = 160000;

typedef __attribute__((ext_vector_type(16))) __bf16 v16bf;
typedef __attribute__((ext_vector_type(8)))  float  v8f;

// ---------------- float <-> order-preserving uint (for atomic segment max) ----
__device__ __forceinline__ unsigned f2ord(float f) {
    unsigned u = __float_as_uint(f);
    return (u & 0x80000000u) ? ~u : (u | 0x80000000u);
}
__device__ __forceinline__ float ord2f(unsigned u) {
    return (u & 0x80000000u) ? __uint_as_float(u ^ 0x80000000u)
                             : __uint_as_float(~u);
}

// ---------------- WMMA GEMM: C[M,Nc] = A[M,K] * B[K,Nc], all f32 in memory ----
// One wave per 16x64 C tile (4 accumulators). K is a compile-time constant so
// the K loop fully unrolls -> 16 v_wmma_f32_16x16x32_bf16 per wave, with the
// A fragment loaded once (four float4) per k-step and reused across the 4
// column tiles. f32->bf16 conversion on the fly, f32 accumulation.
// Fragment layout per CDNA5 ISA 7.12.2:
//   A 16x32 bf16: lanes 0-15 row M=lane, elems 0..7 -> K=kb..kb+7,
//                 elems 8..15 -> K=kb+16..kb+23, with kb = k0 + 8*(lane>=16).
//   B 32x16 bf16: symmetric with N <-> M (lane holds one B column).
//   C/D f32: VGPR v, lanes 0-15 -> (M=v, N=lane); lanes 16-31 -> (M=v+8).
template <int K>
__global__ void k_gemm_wmma(const float* __restrict__ A,
                            const float* __restrict__ B,
                            float* __restrict__ C, int Nc) {
    const int lane = threadIdx.x;
    const int half = lane >> 4;
    const int lr   = lane & 15;
    const int row0 = blockIdx.x * 16;
    const int col0 = blockIdx.y * 64;
    const int arow = row0 + lr;

    v8f acc[4] = {{}, {}, {}, {}};

#pragma unroll
    for (int k0 = 0; k0 < K; k0 += 32) {
        const int kb = k0 + half * 8;

        // A fragment: one row per lane, two 32B-aligned float4 pairs.
        const float* Ap = A + (size_t)arow * K + kb;
        float4 a0 = *(const float4*)(Ap);
        float4 a1 = *(const float4*)(Ap + 4);
        float4 a2 = *(const float4*)(Ap + 16);
        float4 a3 = *(const float4*)(Ap + 20);
        v16bf a;
        a[0] = (__bf16)a0.x;  a[1] = (__bf16)a0.y;
        a[2] = (__bf16)a0.z;  a[3] = (__bf16)a0.w;
        a[4] = (__bf16)a1.x;  a[5] = (__bf16)a1.y;
        a[6] = (__bf16)a1.z;  a[7] = (__bf16)a1.w;
        a[8] = (__bf16)a2.x;  a[9] = (__bf16)a2.y;
        a[10] = (__bf16)a2.z; a[11] = (__bf16)a2.w;
        a[12] = (__bf16)a3.x; a[13] = (__bf16)a3.y;
        a[14] = (__bf16)a3.z; a[15] = (__bf16)a3.w;

#pragma unroll
        for (int t = 0; t < 4; ++t) {
            const int bcol = col0 + t * 16 + lr;
            const float* Bp = B + (size_t)kb * Nc + bcol;
            v16bf b;
#pragma unroll
            for (int j = 0; j < 8; ++j) {
                b[j]     = (__bf16)Bp[(size_t)j * Nc];
                b[8 + j] = (__bf16)Bp[(size_t)(16 + j) * Nc];
            }
            acc[t] = __builtin_amdgcn_wmma_f32_16x16x32_bf16(
                /*neg_a=*/false, a, /*neg_b=*/false, b,
                /*c_mod=*/(short)0, acc[t], /*reuse_a=*/false, /*reuse_b=*/false);
        }
    }

#pragma unroll
    for (int t = 0; t < 4; ++t) {
#pragma unroll
        for (int v = 0; v < 8; ++v) {
            const int r = row0 + v + half * 8;
            C[(size_t)r * Nc + col0 + t * 16 + lr] = acc[t][v];
        }
    }
}

// ---------------- collapse We[64,H,128] . ae[H,128] -> wae[64,H] --------------
__global__ void k_collapse_ae(const float* __restrict__ We,
                              const float* __restrict__ ae,
                              float* __restrict__ wae) {
    int i = blockIdx.x * blockDim.x + threadIdx.x;
    if (i >= 64 * HEADS) return;
    int ii = i >> 2, h = i & 3;
    const float* w = We + ((size_t)ii * HEADS + h) * 128;
    const float* a = ae + (size_t)h * 128;
    float s = 0.f;
    for (int d = 0; d < 128; ++d) s += w[d] * a[d];
    wae[ii * HEADS + h] = s;
}

// ---------------- eproj[E,H] = edge_feats[E,64] @ wae[64,H] -------------------
__global__ void k_eproj(const float* __restrict__ edge,
                        const float* __restrict__ wae,
                        float* __restrict__ eproj) {
    int i = blockIdx.x * blockDim.x + threadIdx.x;
    if (i >= EE * HEADS) return;
    int ed = i >> 2, h = i & 3;
    const float* e = edge + (size_t)ed * 64;
    float s = 0.f;
#pragma unroll 8
    for (int j = 0; j < 64; ++j) s += e[j] * wae[j * HEADS + h];
    eproj[i] = s;
}

// ---------------- el/er[N,H] from feat[N,H,D] and al/ar[H,D] ------------------
__global__ void k_node_logits(const float* __restrict__ feat,
                              const float* __restrict__ al,
                              const float* __restrict__ ar,
                              float* __restrict__ el, float* __restrict__ er,
                              int D) {
    int i = blockIdx.x * blockDim.x + threadIdx.x;
    if (i >= NN * HEADS) return;
    int h = i & 3;
    const float* f = feat + (size_t)i * D;
    const float* a = al + (size_t)h * D;
    const float* b = ar + (size_t)h * D;
    float s1 = 0.f, s2 = 0.f;
    for (int d = 0; d < D; ++d) { float v = f[d]; s1 += v * a[d]; s2 += v * b[d]; }
    el[i] = s1;
    er[i] = s2;
}

// ---------------- init mkey to ord(-inf) --------------------------------------
__global__ void k_init_mkey(unsigned* __restrict__ mkey, int n) {
    int i = blockIdx.x * blockDim.x + threadIdx.x;
    if (i < n) mkey[i] = 0x007FFFFFu;  // f2ord(-inf)
}

// ---------------- per-edge logits + leaky ReLU + atomic segment max -----------
__global__ void k_edge_logit(float* __restrict__ e,
                             const float* __restrict__ el,
                             const float* __restrict__ er,
                             const float* __restrict__ eproj,  // may be null
                             const int* __restrict__ src,
                             const int* __restrict__ dst,
                             unsigned* __restrict__ mkey) {
    int i = blockIdx.x * blockDim.x + threadIdx.x;
    if (i >= EE * HEADS) return;
    int ed = i >> 2, h = i & 3;
    float v = el[src[ed] * HEADS + h] + er[dst[ed] * HEADS + h];
    if (eproj) v += eproj[i];
    v = (v > 0.f) ? v : 0.2f * v;  // leaky relu
    e[i] = v;
    atomicMax(&mkey[dst[ed] * HEADS + h], f2ord(v));
}

// ---------------- exp(e - m[dst]) + atomic segment sum (in-place on e) --------
__global__ void k_edge_exp(float* __restrict__ e,
                           const unsigned* __restrict__ mkey,
                           float* __restrict__ den,
                           const int* __restrict__ dst) {
    int i = blockIdx.x * blockDim.x + threadIdx.x;
    if (i >= EE * HEADS) return;
    int ed = i >> 2, h = i & 3;
    float m  = ord2f(mkey[dst[ed] * HEADS + h]);
    float ex = __expf(e[i] - m);
    e[i] = ex;
    atomicAdd(&den[dst[ed] * HEADS + h], ex);
}

// ---------------- scatter: agg[dst] += feat[src] * alpha ----------------------
// One block per edge; alpha per head staged through LDS; coalesced gather of
// the source feature row (hot in the 192MB L2), f32 atomic adds to agg.
__global__ void k_scatter(float* __restrict__ agg,
                          const float* __restrict__ feat,
                          const float* __restrict__ ex,
                          const float* __restrict__ den,
                          const int* __restrict__ src,
                          const int* __restrict__ dst, int D) {
    int ed = blockIdx.x;
    int s = src[ed], d = dst[ed];
    __shared__ float sal[HEADS];
    if (threadIdx.x < HEADS)
        sal[threadIdx.x] =
            ex[ed * HEADS + threadIdx.x] / (den[d * HEADS + threadIdx.x] + 1e-9f);
    __syncthreads();
    const int HD = HEADS * D;
    const float* fsrc = feat + (size_t)s * HD;
    float* adst = agg + (size_t)d * HD;
    for (int t = threadIdx.x; t < HD; t += blockDim.x)
        atomicAdd(&adst[t], fsrc[t] * sal[t / D]);
}

// ---------------- finalize: relu(mean_h(agg) + res) ---------------------------
__global__ void k_finalize(float* __restrict__ out,
                           const float* __restrict__ agg,
                           const float* __restrict__ resb, int D) {
    int i = blockIdx.x * blockDim.x + threadIdx.x;
    if (i >= NN * D) return;
    int n = i / D, d = i - n * D;
    const float* a = agg + (size_t)n * HEADS * D + d;
    float s = (a[0] + a[(size_t)D] + a[(size_t)2 * D] + a[(size_t)3 * D]) * 0.25f;
    s += resb[i];
    out[i] = s > 0.f ? s : 0.f;
}

// =============================================================================
extern "C" void kernel_launch(void* const* d_in, const int* in_sizes, int n_in,
                              void* d_out, int out_size, void* d_ws, size_t ws_size,
                              hipStream_t stream) {
    const float* x0   = (const float*)d_in[0];   // [N,128]
    const float* efea = (const float*)d_in[1];   // [E,64]
    const int*   src  = (const int*)d_in[2];
    const int*   dst  = (const int*)d_in[3];
    const float* W1   = (const float*)d_in[4];   // [128,4,128]
    const float* We1  = (const float*)d_in[5];   // [64,4,128]
    const float* al1  = (const float*)d_in[6];
    const float* ar1  = (const float*)d_in[7];
    const float* ae1  = (const float*)d_in[8];
    const float* res1 = (const float*)d_in[9];
    const float* W2   = (const float*)d_in[10];
    const float* al2  = (const float*)d_in[11];
    const float* ar2  = (const float*)d_in[12];
    const float* res2 = (const float*)d_in[13];
    const float* W3   = (const float*)d_in[14];
    const float* al3  = (const float*)d_in[15];
    const float* ar3  = (const float*)d_in[16];
    const float* res3 = (const float*)d_in[17];
    const float* W4   = (const float*)d_in[18];  // [128,4,256]
    const float* al4  = (const float*)d_in[19];
    const float* ar4  = (const float*)d_in[20];
    const float* res4 = (const float*)d_in[21];

    // ---- workspace carving (floats), 256B aligned chunks ----
    float* ws = (float*)d_ws;
    size_t off = 0;
    auto carve = [&](size_t nfloats) {
        float* p = ws + off;
        off += (nfloats + 63) & ~(size_t)63;
        return p;
    };
    float*    feat  = carve((size_t)NN * 1024);      // [N, H*Dmax]
    float*    agg   = carve((size_t)NN * 1024);      // [N, H*Dmax]
    float*    resb  = carve((size_t)NN * 256);       // [N, Dmax]
    float*    h0    = carve((size_t)NN * 128);
    float*    h1    = carve((size_t)NN * 128);
    float*    el    = carve((size_t)NN * HEADS);
    float*    er    = carve((size_t)NN * HEADS);
    float*    ebuf  = carve((size_t)EE * HEADS);     // logits, then exp in place
    float*    eproj = carve((size_t)EE * HEADS);
    unsigned* mkey  = (unsigned*)carve((size_t)NN * HEADS);
    float*    den   = carve((size_t)NN * HEADS);
    float*    wae   = carve(64 * HEADS);
    (void)ws_size; (void)in_sizes; (void)n_in; (void)out_size;

    const int TB = 256;
    const int gNH = (NN * HEADS + TB - 1) / TB;
    const int gEH = (EE * HEADS + TB - 1) / TB;

    auto run_layer = [&](const float* xin, int D,
                         const float* W, const float* al, const float* ar,
                         const float* resW, const float* ep, float* hout) {
        const int HD = HEADS * D;
        // feat = x @ W  (WMMA bf16, f32 acc); 16x64 tiles per wave
        dim3 gg(NN / 16, HD / 64);
        k_gemm_wmma<128><<<gg, 32, 0, stream>>>(xin, W, feat, HD);
        // res = x @ resW
        dim3 gr(NN / 16, D / 64);
        k_gemm_wmma<128><<<gr, 32, 0, stream>>>(xin, resW, resb, D);
        // attention logit pieces
        k_node_logits<<<gNH, TB, 0, stream>>>(feat, al, ar, el, er, D);
        // softmax state
        hipMemsetAsync(den, 0, (size_t)NN * HEADS * sizeof(float), stream);
        hipMemsetAsync(agg, 0, (size_t)NN * HD * sizeof(float), stream);
        k_init_mkey<<<gNH, TB, 0, stream>>>(mkey, NN * HEADS);
        // edge softmax
        k_edge_logit<<<gEH, TB, 0, stream>>>(ebuf, el, er, ep, src, dst, mkey);
        k_edge_exp<<<gEH, TB, 0, stream>>>(ebuf, mkey, den, dst);
        // aggregate messages
        k_scatter<<<EE, TB, 0, stream>>>(agg, feat, ebuf, den, src, dst, D);
        // head-mean + residual + relu
        k_finalize<<<(NN * D + TB - 1) / TB, TB, 0, stream>>>(hout, agg, resb, D);
    };

    // Layer 1: fold edge-feature attention term (edge @ (We . ae))
    k_collapse_ae<<<1, 256, 0, stream>>>(We1, ae1, wae);
    k_eproj<<<gEH, TB, 0, stream>>>(efea, wae, eproj);
    run_layer(x0, 128, W1, al1, ar1, res1, eproj, h0);
    // Layers 2,3
    run_layer(h0, 128, W2, al2, ar2, res2, nullptr, h1);
    run_layer(h1, 128, W3, al3, ar3, res3, nullptr, h0);
    // Layer 4 -> d_out directly
    run_layer(h0, 256, W4, al4, ar4, res4, nullptr, (float*)d_out);
}
